// MultiHeadAttention_65034394796539
// MI455X (gfx1250) — compile-verified, matrix operand order checked by reading, not run on hardware
//
#include <hip/hip_runtime.h>
#include <math.h>

// ---------------------------------------------------------------------------
// Problem constants (from reference)
// ---------------------------------------------------------------------------
static constexpr int B  = 4;
static constexpr int S  = 2048;
static constexpr int DM = 1024;
static constexpr int H  = 16;
static constexpr int DH = 64;
static constexpr int HD = H * DH;           // 1024

typedef __bf16 bf16;
typedef __attribute__((ext_vector_type(16))) __bf16 v16bf;
typedef __attribute__((ext_vector_type(8)))  __bf16 v8bf;
typedef __attribute__((ext_vector_type(8)))  float  v8f;

// ---------------------------------------------------------------------------
// CDNA5 async copy (ASYNCcnt-tracked): 16 bytes/lane global -> LDS.
// the LDS address comes from the vdst VGPR; the 32-bit LDS byte offset is the low
// 32 bits of the generic LDS pointer (flat LDS aperture keeps offset in LSBs).
// ---------------------------------------------------------------------------
__device__ __forceinline__ void async_copy_b128(uint32_t lds_byte_off, const void* gaddr) {
    asm volatile("global_load_async_to_lds_b128 %0, %1, off"
                 :: "v"(lds_byte_off), "v"(gaddr)
                 : "memory");
}
#define WAIT_ASYNCCNT_IMM(n) asm volatile("s_wait_asynccnt " #n ::: "memory")

// ---------------------------------------------------------------------------
// Fragment helpers.
//
// CDNA5 16-bit A-fragment layout (16x32, MxK): lane L holds row M = L%16.
// With hi = L/16: vector elements e=0..7  hold K = hi*8      + e,
//                 vector elements e=8..15 hold K = 16 + hi*8 + (e-8).
// B-fragments (32x16, KxN) mirror this: lane holds column N = L%16 with the
// identical K pattern. Both reduce to two contiguous 16-byte loads per lane.
// ---------------------------------------------------------------------------
__device__ __forceinline__ v16bf load_frag_contig(const bf16* p, int hi) {
    const v8bf lo  = *reinterpret_cast<const v8bf*>(p + hi * 8);
    const v8bf hi8 = *reinterpret_cast<const v8bf*>(p + 16 + hi * 8);
    v16bf r;
#pragma unroll
    for (int i = 0; i < 8; ++i) { r[i] = lo[i]; r[i + 8] = hi8[i]; }
    return r;
}

__device__ __forceinline__ v8f wmma_bf16(v16bf a, v16bf b, v8f c) {
    // D(f32 16x16) = A(bf16 16x32) * B(bf16 32x16) + C
    return __builtin_amdgcn_wmma_f32_16x16x32_bf16(
        /*neg_a=*/false, a, /*neg_b=*/false, b,
        /*c_mod=*/(short)0, c, /*reuse_a=*/false, /*reuse_b=*/false);
}

// ---------------------------------------------------------------------------
// Stage 0a: fp32 -> bf16 convert (embed)
// ---------------------------------------------------------------------------
__global__ void mha_cvt_bf16(const float* __restrict__ in, bf16* __restrict__ out, int n) {
    int i = blockIdx.x * blockDim.x + threadIdx.x;
    if (i < n) out[i] = (bf16)in[i];
}

// ---------------------------------------------------------------------------
// Stage 0b: weight transpose+convert: w[h][m][d] -> wT[h][d][m]
// (so B-fragment columns become contiguous K-runs)
// ---------------------------------------------------------------------------
__global__ void mha_cvt_wT(const float* __restrict__ w, bf16* __restrict__ wT,
                           int nh, int M, int D) {
    int i = blockIdx.x * blockDim.x + threadIdx.x;
    int total = nh * M * D;
    if (i >= total) return;
    int h = i / (M * D);
    int r = i % (M * D);
    int m = r / D;
    int d = r % D;
    wT[((size_t)h * D + d) * M + m] = (bf16)w[i];
}

// ---------------------------------------------------------------------------
// Stage 1: per-head projection GEMM.  One wave computes a 32x64 tile
// (two 16-row A fragments share every B fragment -> 8 WMMAs per 12 loads).
// transposeStore==0 -> dst[b,h,s,d]   (Q, K)
// transposeStore==1 -> dst[b,h,d,s]   (V, so PV B-fragments are contiguous)
// ---------------------------------------------------------------------------
__global__ void mha_proj_gemm(const bf16* __restrict__ embB,
                              const bf16* __restrict__ wT,   // [H][DH][DM]
                              bf16* __restrict__ dst,
                              int transposeStore) {
    const int lane = threadIdx.x;
    const int hi = lane >> 4, ln = lane & 15;
    const int s0 = blockIdx.x * 32;       // flattened over B*S
    const int h  = blockIdx.y;

    v8f acc[2][4] = {};
    const bf16* arow0 = embB + (size_t)(s0 + ln) * DM;
    const bf16* arow1 = embB + (size_t)(s0 + 16 + ln) * DM;
    const bf16* wbase = wT + (size_t)h * DH * DM;

    for (int kk = 0; kk < DM; kk += 32) {
        v16bf a0 = load_frag_contig(arow0 + kk, hi);
        v16bf a1 = load_frag_contig(arow1 + kk, hi);
#pragma unroll
        for (int i = 0; i < 4; ++i) {
            v16bf b = load_frag_contig(wbase + (size_t)(i * 16 + ln) * DM + kk, hi);
            acc[0][i] = wmma_bf16(a0, b, acc[0][i]);
            acc[1][i] = wmma_bf16(a1, b, acc[1][i]);
        }
    }

    const int bidx  = s0 / S;             // whole tile stays in one batch (S%32==0)
    const int sbase = s0 % S;
    const size_t bh = (size_t)bidx * H + h;
#pragma unroll
    for (int r = 0; r < 2; ++r) {
#pragma unroll
        for (int i = 0; i < 4; ++i) {
#pragma unroll
            for (int j = 0; j < 8; ++j) {
                const int m = r * 16 + j + hi * 8;   // C/D layout: vgpr j, half hi
                const int d = i * 16 + ln;           // col = lane % 16
                const bf16 v = (bf16)acc[r][i][j];
                if (!transposeStore)
                    dst[(bh * S + sbase + m) * DH + d] = v;
                else
                    dst[(bh * DH + d) * (size_t)S + sbase + m] = v;
            }
        }
    }
}

// ---------------------------------------------------------------------------
// Stage 2: fused causal flash attention.  One wave owns one 16-row query
// tile of one (b,h); streams 32-key blocks with online softmax.  K/V blocks
// are double-buffered in LDS via global_load_async_to_lds_b128 (ASYNCcnt),
// overlapping HBM/L2 latency with softmax + WMMA work.
// ---------------------------------------------------------------------------
__global__ void mha_flash_attn(const bf16* __restrict__ Q,   // [b,h,s,d]
                               const bf16* __restrict__ K,   // [b,h,s,d]
                               const bf16* __restrict__ Vt,  // [b,h,d,s]
                               bf16* __restrict__ O) {       // [b,s,h*DH]
    __shared__ alignas(16) bf16 kTile[2][32 * DH];   // [key][d]   4 KB each
    __shared__ alignas(16) bf16 vTile[2][DH * 32];   // [d][key]   4 KB each
    __shared__ alignas(16) bf16 pLds[16 * 32];

    const int lane = threadIdx.x;
    const int hi = lane >> 4, ln = lane & 15;
    const int s0 = blockIdx.x * 16;
    const int bh = blockIdx.y;            // b*H + h

    const bf16* kb  = K  + (size_t)bh * S * DH;
    const bf16* vtb = Vt + (size_t)bh * DH * S;

    // Stage K block [t0, t0+32): contiguous 4 KB -> 8 async b128 per lane set.
    auto stageK = [&](int t0, int buf) {
        const char* g = (const char*)(kb + (size_t)t0 * DH);
        const uint32_t l = (uint32_t)(uintptr_t)&kTile[buf][0];
        const uint32_t o = (uint32_t)lane * 16;
#pragma unroll
        for (int i = 0; i < 8; ++i)
            async_copy_b128(l + o + i * 512, g + o + i * 512);
    };
    // Stage V block: 64 rows (d), 64 bytes each (row stride S in global).
    auto stageV = [&](int t0, int buf) {
        const uint32_t l = (uint32_t)(uintptr_t)&vTile[buf][0];
        const int r  = lane >> 2;         // 0..7
        const int cb = (lane & 3) * 16;   // byte column within the 64B row
#pragma unroll
        for (int i = 0; i < 8; ++i) {
            const int row = i * 8 + r;
            async_copy_b128(l + (uint32_t)(row * 64 + cb),
                            (const char*)(vtb + (size_t)row * S + t0) + cb);
        }
    };

    const bf16* qbase = Q + ((size_t)bh * S + s0 + ln) * DH;
    const v16bf qa0 = load_frag_contig(qbase, hi);        // d 0..31
    const v16bf qa1 = load_frag_contig(qbase + 32, hi);   // d 32..63

    v8f acc[4] = {};
    float mrow[8], lrow[8];
#pragma unroll
    for (int j = 0; j < 8; ++j) { mrow[j] = -INFINITY; lrow[j] = 0.0f; }

    const float scale = 0.125f;   // 1/sqrt(DH)
    const int nblk = (s0 >> 5) + 1;       // key blocks of 32 up to the diagonal

    stageK(0, 0);
    stageV(0, 0);

    for (int blk = 0; blk < nblk; ++blk) {
        const int t0  = blk * 32;
        const int buf = blk & 1;

        if (blk + 1 < nblk) {
            stageK(t0 + 32, buf ^ 1);
            stageV(t0 + 32, buf ^ 1);
            // 16 outstanding ops belong to the prefetched block; async ops
            // complete in order, so <=16 guarantees the current block landed.
            WAIT_ASYNCCNT_IMM(16);
        } else {
            WAIT_ASYNCCNT_IMM(0);
        }

        // ---- logits: two 16x16 tiles over keys [t0, t0+32), from LDS ----
        v8f c0 = {}, c1 = {};
        {
            const bf16* k0 = &kTile[buf][(size_t)ln * DH];
            c0 = wmma_bf16(qa0, load_frag_contig(k0, hi), c0);
            c0 = wmma_bf16(qa1, load_frag_contig(k0 + 32, hi), c0);
            const bf16* k1 = &kTile[buf][(size_t)(16 + ln) * DH];
            c1 = wmma_bf16(qa0, load_frag_contig(k1, hi), c1);
            c1 = wmma_bf16(qa1, load_frag_contig(k1 + 32, hi), c1);
        }

        // ---- causal mask + online softmax (rows live in 16-lane halves) ----
        float p0[8], p1[8];
#pragma unroll
        for (int j = 0; j < 8; ++j) {
            const int m  = s0 + j + hi * 8;
            const int n0 = t0 + ln;
            const int n1 = t0 + 16 + ln;
            const float l0 = (n0 <= m) ? c0[j] * scale : -INFINITY;
            const float l1 = (n1 <= m) ? c1[j] * scale : -INFINITY;
            p0[j] = l0; p1[j] = l1;

            float mx = fmaxf(l0, l1);
            mx = fmaxf(mx, __shfl_xor(mx, 1));
            mx = fmaxf(mx, __shfl_xor(mx, 2));
            mx = fmaxf(mx, __shfl_xor(mx, 4));
            mx = fmaxf(mx, __shfl_xor(mx, 8));
            const float nm = fmaxf(mrow[j], mx);

            const float alpha = expf(mrow[j] - nm);
            mrow[j] = nm;
            const float e0 = expf(p0[j] - nm);
            const float e1 = expf(p1[j] - nm);
            p0[j] = e0; p1[j] = e1;

            float rs = e0 + e1;
            rs += __shfl_xor(rs, 1);
            rs += __shfl_xor(rs, 2);
            rs += __shfl_xor(rs, 4);
            rs += __shfl_xor(rs, 8);
            lrow[j] = lrow[j] * alpha + rs;

#pragma unroll
            for (int i = 0; i < 4; ++i) acc[i][j] *= alpha;
        }

        // ---- relayout P (C/D layout) -> A-fragment layout through LDS ----
        __syncthreads();
#pragma unroll
        for (int j = 0; j < 8; ++j) {
            const int m = j + hi * 8;
            pLds[m * 32 + ln]      = (bf16)p0[j];
            pLds[m * 32 + 16 + ln] = (bf16)p1[j];
        }
        __syncthreads();
        const v16bf pa = load_frag_contig(pLds + (size_t)ln * 32, hi);

        // ---- acc += P(16x32) * V(32x64), V tile resident in LDS [d][key] ----
#pragma unroll
        for (int i = 0; i < 4; ++i) {
            const bf16* vrow = &vTile[buf][(size_t)(i * 16 + ln) * 32];
            acc[i] = wmma_bf16(pa, load_frag_contig(vrow, hi), acc[i]);
        }
    }

    // ---- epilogue: normalize, write heads-concatenated layout [b,s,h*DH] ----
    const int bidx = bh / H;
    const int h    = bh % H;
#pragma unroll
    for (int i = 0; i < 4; ++i) {
#pragma unroll
        for (int j = 0; j < 8; ++j) {
            const int m = j + hi * 8;
            const int d = i * 16 + ln;
            const float v = acc[i][j] / lrow[j];
            O[((size_t)bidx * S + s0 + m) * HD + h * DH + d] = (bf16)v;
        }
    }
}

// ---------------------------------------------------------------------------
// Stage 3: output projection  out[B*S,1024] = O[B*S,1024] * w_o[1024,1024]
// (w_o pre-transposed to [out][in]); 32x64 tile per wave; fp32 to d_out.
// ---------------------------------------------------------------------------
__global__ void mha_out_proj(const bf16* __restrict__ Obuf,
                             const bf16* __restrict__ woT,  // [DM][HD]
                             float* __restrict__ out) {
    const int lane = threadIdx.x;
    const int hi = lane >> 4, ln = lane & 15;
    const int s0 = blockIdx.x * 32;
    const int n0 = blockIdx.y * 64;

    v8f acc[2][4] = {};
    const bf16* arow0 = Obuf + (size_t)(s0 + ln) * HD;
    const bf16* arow1 = Obuf + (size_t)(s0 + 16 + ln) * HD;
    for (int kk = 0; kk < HD; kk += 32) {
        v16bf a0 = load_frag_contig(arow0 + kk, hi);
        v16bf a1 = load_frag_contig(arow1 + kk, hi);
#pragma unroll
        for (int i = 0; i < 4; ++i) {
            v16bf b = load_frag_contig(woT + (size_t)(n0 + i * 16 + ln) * HD + kk, hi);
            acc[0][i] = wmma_bf16(a0, b, acc[0][i]);
            acc[1][i] = wmma_bf16(a1, b, acc[1][i]);
        }
    }
#pragma unroll
    for (int r = 0; r < 2; ++r) {
#pragma unroll
        for (int i = 0; i < 4; ++i) {
#pragma unroll
            for (int j = 0; j < 8; ++j) {
                const int m = r * 16 + j + hi * 8;
                out[(size_t)(s0 + m) * DM + n0 + i * 16 + ln] = acc[r][i][j];
            }
        }
    }
}

// ---------------------------------------------------------------------------
// Host-side orchestration
// ---------------------------------------------------------------------------
extern "C" void kernel_launch(void* const* d_in, const int* in_sizes, int n_in,
                              void* d_out, int out_size, void* d_ws, size_t ws_size,
                              hipStream_t stream) {
    (void)in_sizes; (void)n_in; (void)out_size; (void)ws_size;

    const float* embed = (const float*)d_in[0];
    // d_in[1] = causal mask (bool) — computed analytically, not read.
    const float* w_q = (const float*)d_in[2];
    const float* w_k = (const float*)d_in[3];
    const float* w_v = (const float*)d_in[4];
    const float* w_o = (const float*)d_in[5];
    float* out = (float*)d_out;

    // ---- workspace carve-up (bf16 element counts) ----
    constexpr size_t N_EMB = (size_t)B * S * DM;      // 8,388,608
    constexpr size_t N_W   = (size_t)H * DM * DH;     // 1,048,576
    constexpr size_t N_WO  = (size_t)HD * DM;         // 1,048,576
    constexpr size_t N_QKV = (size_t)B * H * S * DH;  // 8,388,608
    constexpr size_t N_O   = (size_t)B * S * HD;      // 8,388,608

    bf16* p = (bf16*)d_ws;
    bf16* embB = p;            p += N_EMB;
    bf16* wqT  = p;            p += N_W;
    bf16* wkT  = p;            p += N_W;
    bf16* wvT  = p;            p += N_W;
    bf16* woT  = p;            p += N_WO;
    bf16* Qb   = p;            p += N_QKV;
    bf16* Kb   = p;            p += N_QKV;
    bf16* Vtb  = p;            p += N_QKV;
    bf16* Ob   = p;            p += N_O;

    // Stage 0: converts / transposes
    {
        int n = (int)N_EMB;
        mha_cvt_bf16<<<(n + 255) / 256, 256, 0, stream>>>(embed, embB, n);
        int nw = (int)N_W;
        mha_cvt_wT<<<(nw + 255) / 256, 256, 0, stream>>>(w_q, wqT, H, DM, DH);
        mha_cvt_wT<<<(nw + 255) / 256, 256, 0, stream>>>(w_k, wkT, H, DM, DH);
        mha_cvt_wT<<<(nw + 255) / 256, 256, 0, stream>>>(w_v, wvT, H, DM, DH);
        int no = (int)N_WO;
        mha_cvt_wT<<<(no + 255) / 256, 256, 0, stream>>>(w_o, woT, 1, HD, DM);
    }

    // Stage 1: QKV projections (one wave per 32x64 tile)
    {
        dim3 grid(B * S / 32, H);
        mha_proj_gemm<<<grid, 32, 0, stream>>>(embB, wqT, Qb,  0);
        mha_proj_gemm<<<grid, 32, 0, stream>>>(embB, wkT, Kb,  0);
        mha_proj_gemm<<<grid, 32, 0, stream>>>(embB, wvT, Vtb, 1);
    }

    // Stage 2: fused causal flash attention (async double-buffered K/V)
    {
        dim3 grid(S / 16, B * H);
        mha_flash_attn<<<grid, 32, 0, stream>>>(Qb, Kb, Vtb, Ob);
    }

    // Stage 3: output projection -> fp32 d_out
    {
        dim3 grid(B * S / 32, DM / 64);
        mha_out_proj<<<grid, 32, 0, stream>>>(Ob, woT, out);
    }
}
